// MultiHeadAttention_70205535420826
// MI455X (gfx1250) — matrix operand order repaired and checked by reading, not compiled, not run on hardware
//
#include <hip/hip_runtime.h>
#include <hip/hip_bf16.h>

#define SEQ 4096
#define DM  1024
#define NH  16
#define DK  64

typedef unsigned short u16;
typedef __attribute__((ext_vector_type(8)))  __bf16 bf16x8;
typedef __attribute__((ext_vector_type(16))) __bf16 bf16x16;
typedef __attribute__((ext_vector_type(8)))  float  f32x8;
typedef __attribute__((ext_vector_type(4)))  unsigned u32x4;
typedef __attribute__((ext_vector_type(8)))  unsigned u32x8;

__device__ __forceinline__ u16 f2bf(float f) {
  union { float f; unsigned u; } x; x.f = f;
  unsigned r = x.u + 0x7FFFu + ((x.u >> 16) & 1u);
  return (u16)(r >> 16);
}

// Build a 16-element bf16 A/B fragment from two 16B chunks (ds_load_b128 x2).
__device__ __forceinline__ bf16x16 frag(const u16* lo, const u16* hi) {
  bf16x8 a = *(const bf16x8*)lo;
  bf16x8 b = *(const bf16x8*)hi;
  return __builtin_shufflevector(a, b, 0,1,2,3,4,5,6,7,8,9,10,11,12,13,14,15);
}

__device__ __forceinline__ f32x8 wmma_bf16(bf16x16 a, bf16x16 b, f32x8 c) {
  return __builtin_amdgcn_wmma_f32_16x16x32_bf16(false, a, false, b, (short)0, c,
                                                 false, false);
}

__device__ __forceinline__ float half16_max(float v) {
  v = fmaxf(v, __shfl_xor(v, 1));
  v = fmaxf(v, __shfl_xor(v, 2));
  v = fmaxf(v, __shfl_xor(v, 4));
  v = fmaxf(v, __shfl_xor(v, 8));
  return v;
}
__device__ __forceinline__ float half16_sum(float v) {
  v += __shfl_xor(v, 1);
  v += __shfl_xor(v, 2);
  v += __shfl_xor(v, 4);
  v += __shfl_xor(v, 8);
  return v;
}

// ---------------------------------------------------------------------------
// TDM: async-load a 64x64 bf16 tile (row stride 64 elems in memory) into LDS
// with rows padded to 72 halfwords (36 dwords) via the D# pad feature.
// D# bit layout per CDNA5 ISA ch.8 (Group0: count/lds_addr/global_addr/type,
// Group1: data_size/pad/dims/tile/strides). Issued by one wave; tracked with
// TENSORcnt.
// ---------------------------------------------------------------------------
__device__ __forceinline__ void tdm_load_tile64x64(const u16* gsrc,
                                                   unsigned lds_byte_addr) {
  unsigned long long ga = (unsigned long long)(uintptr_t)gsrc;
  u32x4 g0;
  g0[0] = 1u;                                   // count=1, user descriptor
  g0[1] = lds_byte_addr;                        // lds_addr
  g0[2] = (unsigned)(ga & 0xFFFFFFFFull);       // global_addr[31:0]
  g0[3] = (unsigned)((ga >> 32) & 0x1FFFFFFull) // global_addr[56:32]
          | (2u << 30);                         // type = 2 ("image")
  u32x8 g1;
  g1[0] = (1u << 16)        // data_size = 2 bytes
        | (1u << 20)        // pad_enable
        | (4u << 22)        // pad_interval: 32 dwords (one 64-elem bf16 row)
        | (3u << 25);       // pad_amount: 4 dwords (16B) -> 144B LDS row
  g1[1] = (64u << 16);      // tensor_dim0 = 64  (bits 79:48, low half)
  g1[2] = (4096u << 16);    // tensor_dim0 hi = 0 | tensor_dim1 = 4096 (low)
  g1[3] = (64u << 16);      // tensor_dim1 hi = 0 | tile_dim0 = 64
  g1[4] = 64u;              // tile_dim1 = 64, tile_dim2 = 0
  g1[5] = 64u;              // tensor_dim0_stride = 64 (low 32)
  g1[6] = 0u;               // stride hi / tensor_dim1_stride low
  g1[7] = 0u;
  u32x4 gz;
  gz[0] = 0u; gz[1] = 0u; gz[2] = 0u; gz[3] = 0u;   // groups 2/3: dims unused
  asm volatile("tensor_load_to_lds %0, %1, %2, %3"
               :: "s"(g0), "s"(g1), "s"(gz), "s"(gz)
               : "memory");
}

// ---------------------------------------------------------------------------
// C = X(MxK) * W(NxK)^T with bf16 WMMA, f32 accumulation.
// MODE 0: store bf16 head-major [n>>6][m][n&63] (projection path)
// MODE 1: store f32 relu(C) + Res (output path)
// ---------------------------------------------------------------------------
template<bool XBF16, int MODE>
__global__ __launch_bounds__(256)
void gemm128x128_wmma(const void* __restrict__ Xv, const float* __restrict__ W,
                      void* __restrict__ Outv, const float* __restrict__ Res,
                      int M, int N, int K)
{
  __shared__ u16 As[128 * 40];   // 128 rows x 32 K, stride 40 (80B, 16B aligned)
  __shared__ u16 Bs[128 * 40];

  const int tid  = threadIdx.x;
  const int lane = tid & 31;
  const int wid  = tid >> 5;
  const int wm   = wid & 3;        // 4 waves along M -> 32-row slab
  const int wn   = wid >> 2;       // 2 waves along N -> 64-col slab
  const int lrow = lane & 15;
  const int hi16 = lane >> 4;
  const int akb  = hi16 ? 8 : 0;   // A frag K-base per ISA layout
  const int bkb  = hi16 ? 8 : 0;   // B frag K-base per ISA layout
  const int m0   = blockIdx.y * 128;
  const int n0   = blockIdx.x * 128;

  const int sr = tid >> 1;          // staging row 0..127
  const int sc = (tid & 1) * 16;    // staging col 0/16

  f32x8 acc[2][4];
  #pragma unroll
  for (int i = 0; i < 2; i++)
    #pragma unroll
    for (int j = 0; j < 4; j++)
      #pragma unroll
      for (int r = 0; r < 8; r++) acc[i][j][r] = 0.f;

  for (int k0 = 0; k0 < K; k0 += 32) {
    // ---- stage A tile ----
    if constexpr (XBF16) {
      const u16* xp = (const u16*)Xv + (size_t)(m0 + sr) * K + k0 + sc;
      uint4 a = ((const uint4*)xp)[0];
      uint4 b = ((const uint4*)xp)[1];
      ((uint4*)&As[sr * 40 + sc])[0] = a;
      ((uint4*)&As[sr * 40 + sc])[1] = b;
    } else {
      const float* xp = (const float*)Xv + (size_t)(m0 + sr) * K + k0 + sc;
      alignas(16) u16 tmp[16];
      #pragma unroll
      for (int i = 0; i < 4; i++) {
        float4 f = ((const float4*)xp)[i];
        tmp[4*i+0] = f2bf(f.x); tmp[4*i+1] = f2bf(f.y);
        tmp[4*i+2] = f2bf(f.z); tmp[4*i+3] = f2bf(f.w);
      }
      ((uint4*)&As[sr * 40 + sc])[0] = ((uint4*)tmp)[0];
      ((uint4*)&As[sr * 40 + sc])[1] = ((uint4*)tmp)[1];
    }
    // ---- stage B tile (rows of W, always fp32) ----
    {
      const float* wp = W + (size_t)(n0 + sr) * K + k0 + sc;
      alignas(16) u16 tmp[16];
      #pragma unroll
      for (int i = 0; i < 4; i++) {
        float4 f = ((const float4*)wp)[i];
        tmp[4*i+0] = f2bf(f.x); tmp[4*i+1] = f2bf(f.y);
        tmp[4*i+2] = f2bf(f.z); tmp[4*i+3] = f2bf(f.w);
      }
      ((uint4*)&Bs[sr * 40 + sc])[0] = ((uint4*)tmp)[0];
      ((uint4*)&Bs[sr * 40 + sc])[1] = ((uint4*)tmp)[1];
    }
    __syncthreads();

    bf16x16 af[2], bfm[4];
    #pragma unroll
    for (int i = 0; i < 2; i++) {
      const u16* p = &As[(wm * 32 + i * 16 + lrow) * 40];
      af[i] = frag(p + akb, p + akb + 16);
    }
    #pragma unroll
    for (int j = 0; j < 4; j++) {
      const u16* p = &Bs[(wn * 64 + j * 16 + lrow) * 40];
      bfm[j] = frag(p + bkb, p + bkb + 16);
    }
    #pragma unroll
    for (int i = 0; i < 2; i++)
      #pragma unroll
      for (int j = 0; j < 4; j++)
        acc[i][j] = wmma_bf16(af[i], bfm[j], acc[i][j]);
    __syncthreads();
  }

  // ---- epilogue ----
  #pragma unroll
  for (int i = 0; i < 2; i++) {
    #pragma unroll
    for (int j = 0; j < 4; j++) {
      #pragma unroll
      for (int r = 0; r < 8; r++) {
        int row = m0 + wm * 32 + i * 16 + r + hi16 * 8;
        int col = n0 + wn * 64 + j * 16 + lrow;
        float v = acc[i][j][r];
        if constexpr (MODE == 0) {
          int h = col >> 6, d = col & 63;
          ((u16*)Outv)[((size_t)h * M + row) * 64 + d] = f2bf(v);
        } else {
          float o = fmaxf(v, 0.f) + Res[(size_t)row * N + col];
          ((float*)Outv)[(size_t)row * N + col] = o;
        }
      }
    }
  }
}

// ---------------------------------------------------------------------------
// Flash attention: per (128-query block, head). 8 waves x 16 queries.
// K tiles double-buffered via TDM tensor_load_to_lds (wave 0 issues, all waves
// sync after s_wait_tensorcnt). V tiles double-buffered manually (transposed
// [d][key] so PV B-fragments are contiguous); V global loads are hoisted above
// the barrier to overlap compute.
// ---------------------------------------------------------------------------
__global__ __launch_bounds__(256)
void flash_attn_wmma(const u16* __restrict__ Qh, const u16* __restrict__ Kh,
                     const u16* __restrict__ Vh, u16* __restrict__ Ao)
{
  __shared__ u16 Kld[2][64 * 72];   // [key][d], stride 72 (144B) -- TDM-filled
  __shared__ u16 Vld[2][64 * 72];   // [d][key] (transposed)
  __shared__ u16 Pld[8 * 16 * 72];  // per-wave P re-layout C->A

  const int tid  = threadIdx.x;
  const int lane = tid & 31;
  const int wid  = tid >> 5;
  const int lrow = lane & 15;
  const int hi16 = lane >> 4;
  const int akb  = hi16 ? 8 : 0;
  const int bkb  = hi16 ? 8 : 0;
  const int h    = blockIdx.y;
  const int q0   = blockIdx.x * 128 + wid * 16;

  const u16* Kbase = Kh + (size_t)h * SEQ * 64;
  const u16* Vbase = Vh + (size_t)h * SEQ * 64;
  const unsigned kldAddr[2] = { (unsigned)(uintptr_t)&Kld[0][0],
                                (unsigned)(uintptr_t)&Kld[1][0] };

  // Q fragments (d = 0..31, 32..63), kept for all 64 KV blocks
  const u16* qp = Qh + ((size_t)h * SEQ + q0 + lrow) * 64;
  bf16x16 qa0 = frag(qp + akb,      qp + akb + 16);
  bf16x16 qa1 = frag(qp + 32 + akb, qp + 32 + akb + 16);

  f32x8 o[4];
  #pragma unroll
  for (int t = 0; t < 4; t++)
    #pragma unroll
    for (int r = 0; r < 8; r++) o[t][r] = 0.f;
  float m_run[8], l_run[8];
  #pragma unroll
  for (int r = 0; r < 8; r++) { m_run[r] = -3.0e38f; l_run[r] = 0.f; }

  u16* pw = &Pld[wid * 16 * 72];
  const float scl = 0.125f;   // 1/sqrt(64)
  const int vkey = tid & 63;
  const int vdv  = (tid >> 6) * 16;

  // ---- prologue: kick TDM for K block 0; stage V block 0 ----
  if (wid == 0) tdm_load_tile64x64(Kbase, kldAddr[0]);
  {
    const u16* vp = Vbase + (size_t)vkey * 64 + vdv;
    alignas(16) u16 tmp[16];
    ((uint4*)tmp)[0] = ((const uint4*)vp)[0];
    ((uint4*)tmp)[1] = ((const uint4*)vp)[1];
    #pragma unroll
    for (int e = 0; e < 16; e++) Vld[0][(vdv + e) * 72 + vkey] = tmp[e];
  }

  const int nblk = SEQ / 64;
  for (int i = 0; i < nblk; i++) {
    const int b = i & 1;
    // issue next K DMA into the free buffer; wait for current buffer (in-order)
    if (wid == 0) {
      if (i + 1 < nblk) {
        tdm_load_tile64x64(Kbase + (size_t)(i + 1) * 64 * 64, kldAddr[b ^ 1]);
        __builtin_amdgcn_s_wait_tensorcnt(1);
      } else {
        __builtin_amdgcn_s_wait_tensorcnt(0);
      }
    }
    // prefetch next V block into registers (overlaps barrier + compute)
    uint4 va0, va1;
    if (i + 1 < nblk) {
      const u16* vp = Vbase + (size_t)(i + 1) * 64 * 64 + (size_t)vkey * 64 + vdv;
      va0 = ((const uint4*)vp)[0];
      va1 = ((const uint4*)vp)[1];
    }
    __syncthreads();   // K(b) DMA landed, V(b) stores visible

    // S = Q * K^T  (4 score tiles of 16x16, each K-reduced over d=64)
    f32x8 s[4];
    #pragma unroll
    for (int j = 0; j < 4; j++)
      #pragma unroll
      for (int r = 0; r < 8; r++) s[j][r] = 0.f;
    #pragma unroll
    for (int j = 0; j < 4; j++) {
      const u16* kp = &Kld[b][(j * 16 + lrow) * 72];
      bf16x16 b0 = frag(kp + bkb,      kp + bkb + 16);
      bf16x16 b1 = frag(kp + 32 + bkb, kp + 32 + bkb + 16);
      s[j] = wmma_bf16(qa0, b0, s[j]);
      s[j] = wmma_bf16(qa1, b1, s[j]);
    }

    // online softmax (row lives across 16 lanes of one half-wave)
    float mnew[8], alpha[8], psum[8];
    #pragma unroll
    for (int r = 0; r < 8; r++) {
      float v = fmaxf(fmaxf(s[0][r], s[1][r]), fmaxf(s[2][r], s[3][r])) * scl;
      v = half16_max(v);
      mnew[r]  = fmaxf(m_run[r], v);
      alpha[r] = __expf(m_run[r] - mnew[r]);
      m_run[r] = mnew[r];
      psum[r]  = 0.f;
    }
    #pragma unroll
    for (int j = 0; j < 4; j++)
      #pragma unroll
      for (int r = 0; r < 8; r++) {
        float p = __expf(s[j][r] * scl - mnew[r]);
        psum[r] += p;
        pw[(r + hi16 * 8) * 72 + j * 16 + lrow] = f2bf(p);  // C-layout -> [q][k]
      }
    #pragma unroll
    for (int r = 0; r < 8; r++)
      l_run[r] = l_run[r] * alpha[r] + half16_sum(psum[r]);
    #pragma unroll
    for (int t = 0; t < 4; t++)
      #pragma unroll
      for (int r = 0; r < 8; r++) o[t][r] *= alpha[r];

    // O += P * V  (P A-fragments from per-wave LDS, V^T B-fragments)
    const u16* pr = pw + lrow * 72;
    bf16x16 pa0 = frag(pr + akb,      pr + akb + 16);
    bf16x16 pa1 = frag(pr + 32 + akb, pr + 32 + akb + 16);
    #pragma unroll
    for (int t = 0; t < 4; t++) {
      const u16* vp2 = &Vld[b][(t * 16 + lrow) * 72];
      bf16x16 vb0 = frag(vp2 + bkb,      vp2 + bkb + 16);
      bf16x16 vb1 = frag(vp2 + 32 + bkb, vp2 + 32 + bkb + 16);
      o[t] = wmma_bf16(pa0, vb0, o[t]);
      o[t] = wmma_bf16(pa1, vb1, o[t]);
    }

    // scatter prefetched V block (transposed) into the free buffer
    if (i + 1 < nblk) {
      alignas(16) u16 tmp[16];
      ((uint4*)tmp)[0] = va0;
      ((uint4*)tmp)[1] = va1;
      #pragma unroll
      for (int e = 0; e < 16; e++) Vld[b ^ 1][(vdv + e) * 72 + vkey] = tmp[e];
    }
    __syncthreads();   // all reads of buffer b done -> reusable next iter
  }

  // normalize and store concat-layout bf16 [seq][h*64+d]
  #pragma unroll
  for (int r = 0; r < 8; r++) {
    float inv = 1.0f / l_run[r];
    int q = q0 + r + hi16 * 8;
    #pragma unroll
    for (int t = 0; t < 4; t++) {
      int d = t * 16 + lrow;
      Ao[(size_t)q * DM + h * 64 + d] = f2bf(o[t][r] * inv);
    }
  }
}

// ---------------------------------------------------------------------------
extern "C" void kernel_launch(void* const* d_in, const int* in_sizes, int n_in,
                              void* d_out, int out_size, void* d_ws, size_t ws_size,
                              hipStream_t stream)
{
  (void)in_sizes; (void)n_in; (void)out_size; (void)ws_size;
  const float* q   = (const float*)d_in[0];
  const float* k   = (const float*)d_in[1];
  const float* v   = (const float*)d_in[2];
  const float* Wq  = (const float*)d_in[3];
  const float* Wk  = (const float*)d_in[4];
  const float* Wv  = (const float*)d_in[5];
  const float* Wfc = (const float*)d_in[6];

  const size_t per = (size_t)NH * SEQ * DK;   // 4,194,304 bf16 elements
  u16* Qh = (u16*)d_ws;
  u16* Kh = Qh + per;
  u16* Vh = Kh + per;
  u16* Ao = Vh + per;

  dim3 blk(256);
  dim3 gg(DM / 128, SEQ / 128);

  gemm128x128_wmma<false, 0><<<gg, blk, 0, stream>>>(q, Wq, Qh, nullptr, SEQ, DM, DM);
  gemm128x128_wmma<false, 0><<<gg, blk, 0, stream>>>(k, Wk, Kh, nullptr, SEQ, DM, DM);
  gemm128x128_wmma<false, 0><<<gg, blk, 0, stream>>>(v, Wv, Vh, nullptr, SEQ, DM, DM);

  flash_attn_wmma<<<dim3(SEQ / 128, NH), blk, 0, stream>>>(Qh, Kh, Vh, Ao);

  gemm128x128_wmma<true, 1><<<gg, blk, 0, stream>>>(Ao, Wfc, (float*)d_out, q,
                                                    SEQ, DM, DM);
}
